// RouteNetOneToOneOutputGroupedInputs_48902497632407
// MI455X (gfx1250) — matrix operand order repaired and checked by reading, not compiled, not run on hardware
//
#include <hip/hip_runtime.h>
#include <hip/hip_bf16.h>

// ---------------- problem constants ----------------
#define LAYERS   10
#define PB       20        // banks per layer
#define DD       10        // neurons per hidden bank
#define NINP     40        // neurons per input group
#define BATCH    16384
#define NEDGES   3600

typedef _Float16 v8h  __attribute__((ext_vector_type(8)));
typedef _Float16 v16h __attribute__((ext_vector_type(16)));
typedef float    v8f  __attribute__((ext_vector_type(8)));

// ---------------- packed weight layout in d_ws (units: _Float16) ----------------
// W_in   : [g=20][N=16][K=64]             B-frag for h0 GEMM (K=n padded 40->64, N=d padded 10->16)
// W_gate : [l=9][s=20][N=32][K=32]        rows K = d + 16*(s&1) hold gw[l,s,t=N,d]; others 0
// W_data : [l=9][sp=10][t=20][N=16][K=32] rows K = d + 16*(s-2sp) hold dw[l,s,t,e=N,d]; others 0
#define WIN_OFF   0
#define WIN_SZ    (20 * 16 * 64)                 // 20480
#define WGATE_OFF (WIN_OFF + WIN_SZ)
#define WGATE_SZ  (9 * 20 * 32 * 32)             // 184320
#define WDATA_OFF (WGATE_OFF + WGATE_SZ)
#define WDATA_SZ  (9 * 10 * 20 * 16 * 32)        // 921600
#define WTOTAL    (WDATA_OFF + WDATA_SZ)         // 1126400 halves  (~2.25 MB)

// ---------------- helpers ----------------
__device__ __forceinline__ v16h make16(v8h lo, v8h hi) {
    return __builtin_shufflevector(lo, hi, 0,1,2,3,4,5,6,7,8,9,10,11,12,13,14,15);
}
__device__ __forceinline__ v16h load_bfrag(const _Float16* p) {   // halves p[0..7] and p[16..23] per lane
    const v8h* q = (const v8h*)p;
    return make16(q[0], q[2]);
}
__device__ __forceinline__ v8f wmma16(v16h a, v16h b, v8f c) {
    return __builtin_amdgcn_wmma_f32_16x16x32_f16(false, a, false, b, (short)0, c, false, false);
}
// CDNA5 async global->LDS copy (ASYNCcnt-tracked); 16 bytes per lane.
__device__ __forceinline__ void async_b128(unsigned int lds_off, unsigned long long gaddr) {
    asm volatile("global_load_async_to_lds_b128 %0, %1, off"
                 :: "v"(lds_off), "v"(gaddr)
                 : "memory");
}
__device__ __forceinline__ void wait_async0() {
    asm volatile("s_wait_asynccnt 0x0" ::: "memory");
}

// ---------------- prep: f32 weights -> fragment-ready f16 packs in d_ws ----------------
__global__ void prep_pack(const float* __restrict__ in_w,
                          const float* __restrict__ gate_w,
                          const float* __restrict__ data_w,
                          _Float16*    __restrict__ ws,
                          float*       __restrict__ d_out) {
    int idx = blockIdx.x * 256 + threadIdx.x;
    if (idx == 0) d_out[BATCH * PB + BATCH] = 0.0f;    // zero prob_open_gate accumulator slot
    if (idx >= WTOTAL) return;
    float v = 0.0f;
    if (idx < WGATE_OFF) {
        int r = idx;
        int g = r / (16 * 64); r %= (16 * 64);
        int N = r / 64, K = r % 64;
        if (N < DD && K < NINP) v = in_w[(g * DD + N) * NINP + K];
    } else if (idx < WDATA_OFF) {
        int r = idx - WGATE_OFF;
        int ls = r / 1024; r %= 1024;
        int N = r / 32, K = r % 32;
        int l = ls / PB, s = ls % PB;
        int d = K & 15;
        if (N < PB && ((K >> 4) == (s & 1)) && d < DD)
            v = gate_w[((l * PB + s) * PB + N) * DD + d];
    } else {
        int r = idx - WDATA_OFF;
        int u = r / 512; r %= 512;
        int N = r / 32, K = r % 32;
        int l = u / 200; int r2 = u % 200;
        int sp = r2 / PB, t = r2 % PB;
        int s = 2 * sp + (K >> 4);
        int d = K & 15;
        if (N < DD && d < DD)
            v = data_w[(((l * PB + s) * PB + t) * DD + N) * DD + d];
    }
    ws[idx] = (_Float16)v;
}

// ---------------- main kernel ----------------
// 64 threads = 2 waves per block; each wave owns 16 batch rows end-to-end.
// Per-wave LDS: h ping-pong 2x(20*16*16 f16) + gate table (2*16*10 f32).
// Block-shared LDS: 10KB async-staged data-weight slice (one (pass,sp) t-range).
__global__ __launch_bounds__(64)
void routenet_fwd(const float* __restrict__ x,
                  const float* __restrict__ gate_b,
                  const float* __restrict__ out_w,
                  const _Float16* __restrict__ ws,
                  float* __restrict__ d_out) {
    __shared__ __align__(32) _Float16 sh_h[2][2][PB * 16 * 16];  // 40KB
    __shared__ __align__(16) float    sh_g[2][2 * 16 * 10];      // 2.5KB
    __shared__ __align__(32) _Float16 sh_stage[10 * 512];        // 10KB, async staging (x aliases it)

    const int wave = threadIdx.x >> 5;
    const int lane = threadIdx.x & 31;
    const int half = lane >> 4;       // K-half selector in fragments
    const int n16  = lane & 15;       // A-row m / B-and-D column N
    const int b0   = blockIdx.x * 32 + wave * 16;   // base batch row of this wave

    _Float16* hb0 = sh_h[wave][0];
    _Float16* hb1 = sh_h[wave][1];
    float*    gl  = sh_g[wave];
    _Float16* xt  = sh_stage + wave * 1024;         // 16x64 f16 x-staging (pre-layer phase only)

    // zero x staging pads (K in [40,64) stays zero)
    for (int k = lane; k < 16 * 64; k += 32) xt[k] = (_Float16)0.0f;

    // -------- layer 0: h0[b,g,d] = relu(x[g,b,:] @ in_w[g]^T) --------
    for (int g = 0; g < PB; ++g) {
        for (int r = 0; r < 16; ++r) {
            const float* xr = x + ((size_t)g * BATCH + (size_t)(b0 + r)) * NINP;
            if (lane < NINP)      xt[r * 64 + lane]      = (_Float16)xr[lane];
            if (lane + 32 < NINP) xt[r * 64 + lane + 32] = (_Float16)xr[lane + 32];
        }
        v8f hc = {0.f, 0.f, 0.f, 0.f, 0.f, 0.f, 0.f, 0.f};
#pragma unroll
        for (int c = 0; c < 2; ++c) {
            v8h lo = *(const v8h*)(xt + n16 * 64 + c * 32 + 8 * half);
            v8h hi = *(const v8h*)(xt + n16 * 64 + c * 32 + 16 + 8 * half);
            v16h a = make16(lo, hi);
            v16h b = load_bfrag(ws + WIN_OFF + (g * 16 + n16) * 64 + c * 32 + 8 * half);
            hc = wmma16(a, b, hc);
        }
#pragma unroll
        for (int j = 0; j < 8; ++j)
            hb0[(g * 16 + j + 8 * half) * 16 + n16] = (_Float16)fmaxf(hc[j], 0.0f);
    }
    __syncthreads();   // xt region is reused as the shared weight staging buffer below

    // -------- layers: gate + gated data transform, all via WMMA --------
    float tg[8] = {0, 0, 0, 0, 0, 0, 0, 0};
    int   cnt = 0;
    const unsigned int stage_base = (unsigned int)(uintptr_t)(void*)sh_stage;

    for (int l = 0; l < LAYERS - 1; ++l) {
        _Float16* hcur = (l & 1) ? hb1 : hb0;
        _Float16* hnxt = (l & 1) ? hb0 : hb1;
        const _Float16* wgl = ws + WGATE_OFF + l * (PB * 1024);
        const _Float16* wdl = ws + WDATA_OFF + l * (10 * PB * 512);

        for (int pass = 0; pass < 2; ++pass) {
            const int tlo = pass * 10;
            v8f acc[10];
#pragma unroll
            for (int tt = 0; tt < 10; ++tt)
                acc[tt] = (v8f){0.f, 0.f, 0.f, 0.f, 0.f, 0.f, 0.f, 0.f};

            for (int sp = 0; sp < 10; ++sp) {
                const int s0 = 2 * sp;

                // ---- kick off async staging of this (pass,sp) data-weight slice (10KB) ----
                {
                    const char* src = (const char*)(wdl + sp * (PB * 512) + tlo * 512);
#pragma unroll
                    for (int k = 0; k < 10; ++k) {
                        unsigned int o = (unsigned int)(threadIdx.x * 16 + k * 1024);
                        async_b128(stage_base + o, (unsigned long long)(uintptr_t)(src + o));
                    }
                }
                if (sp < 9) __builtin_prefetch(wgl + (s0 + 2) * 1024, 0, 1);

                // unscaled A pair: K 0..15 <- h[s0], K 16..31 <- h[s0+1]
                v8h hl0 = *(const v8h*)(hcur + (s0 * 16 + n16) * 16 + 8 * half);
                v8h hl1 = *(const v8h*)(hcur + ((s0 + 1) * 16 + n16) * 16 + 8 * half);
                v16h apair = make16(hl0, hl1);

                // ---- gates (overlap the async copy); per-s B has the other s's K rows zeroed ----
#pragma unroll
                for (int sw = 0; sw < 2; ++sw) {
                    const _Float16* wg = wgl + (s0 + sw) * 1024;
                    const float* gbp = gate_b + ((l * PB + s0 + sw) * PB);
                    v8f z = {0.f, 0.f, 0.f, 0.f, 0.f, 0.f, 0.f, 0.f};
                    // N-chunk 0: t = n16 (0..15)
                    v8f gf = wmma16(apair, load_bfrag(wg + (0 + n16) * 32 + 8 * half), z);
                    float gbv = gbp[n16];
                    bool inr = (n16 >= tlo) && (n16 < tlo + 10);
#pragma unroll
                    for (int j = 0; j < 8; ++j) {
                        float pre = gf[j] + gbv;
                        if (inr) {
                            cnt += (pre > 0.0f) ? 1 : 0;
                            float gt = fminf(fmaxf(pre, 0.0f), 1.0f);
                            tg[j] += gt;
                            gl[(sw * 16 + j + 8 * half) * 10 + (n16 - tlo)] = gt;
                        }
                    }
                    if (pass == 1) {
                        // N-chunk 1: t = 16 + n16 (valid for n16 < 4)
                        v8f gf1 = wmma16(apair, load_bfrag(wg + (16 + n16) * 32 + 8 * half), z);
                        float gbv1 = (n16 < 4) ? gbp[16 + n16] : 0.0f;
#pragma unroll
                        for (int j = 0; j < 8; ++j) {
                            float pre = gf1[j] + gbv1;
                            if (n16 < 4) {
                                cnt += (pre > 0.0f) ? 1 : 0;
                                float gt = fminf(fmaxf(pre, 0.0f), 1.0f);
                                tg[j] += gt;
                                gl[(sw * 16 + j + 8 * half) * 10 + 6 + n16] = gt;
                            }
                        }
                    }
                }

                // ---- staged weights ready: wait own ASYNCcnt, then block barrier ----
                wait_async0();
                __syncthreads();

                // ---- gated data transform: fold gate(b,s,t) into A, WMMA-accumulate ----
#pragma unroll
                for (int tt = 0; tt < 10; ++tt) {
                    _Float16 g0h = (_Float16)gl[(0 * 16 + n16) * 10 + tt];
                    _Float16 g1h = (_Float16)gl[(16 + n16) * 10 + tt];
                    v8h g0v = {g0h, g0h, g0h, g0h, g0h, g0h, g0h, g0h};
                    v8h g1v = {g1h, g1h, g1h, g1h, g1h, g1h, g1h, g1h};
                    v16h as = make16(hl0 * g0v, hl1 * g1v);   // -> v_pk_mul_f16
                    v16h bf = load_bfrag(sh_stage + tt * 512 + n16 * 32 + 8 * half);
                    acc[tt] = wmma16(as, bf, acc[tt]);
                }
                __syncthreads();   // all waves done reading before next slice overwrites
            } // sp

            // write back h_next (relu) for this t-range
#pragma unroll
            for (int tt = 0; tt < 10; ++tt)
#pragma unroll
                for (int j = 0; j < 8; ++j)
                    hnxt[((tlo + tt) * 16 + j + 8 * half) * 16 + n16] =
                        (_Float16)fmaxf(acc[tt][j], 0.0f);
        } // pass
    } // layers

    __syncthreads();
    _Float16* hf = hb1;   // after 9 layers (last l=8 even) result is in hb1

    // -------- output einsum: out[b,p] = h[b,p,:] . out_w[p,:]  (block-diagonal -> VALU) --------
    if (lane < 16) {
        const int b = b0 + n16;
        for (int p = 0; p < PB; ++p) {
            float sum = 0.0f;
#pragma unroll
            for (int d = 0; d < DD; ++d)
                sum += (float)hf[(p * 16 + n16) * 16 + d] * out_w[p * DD + d];
            d_out[b * PB + p] = sum;
        }
    }

    // -------- total_gate_act: reduce tg rows across the 16 t-lanes --------
    float* red = gl;   // reuse (256 of 320 floats)
#pragma unroll
    for (int j = 0; j < 8; ++j)
        red[(j + 8 * half) * 16 + n16] = tg[j];
    __syncthreads();
    if (lane < 16) {
        float s = 0.0f;
#pragma unroll
        for (int k = 0; k < 16; ++k) s += red[lane * 16 + k];
        d_out[BATCH * PB + b0 + lane] = s * (1.0f / (float)NEDGES);
    }

    // -------- prob_open_gate: wave reduce + one float atomic --------
    for (int o = 16; o > 0; o >>= 1) cnt += __shfl_xor(cnt, o, 32);
    if (lane == 0)
        atomicAdd(d_out + BATCH * PB + BATCH,
                  (float)cnt * (1.0f / ((float)NEDGES * (float)BATCH)));
}

// ---------------- launch ----------------
extern "C" void kernel_launch(void* const* d_in, const int* in_sizes, int n_in,
                              void* d_out, int out_size, void* d_ws, size_t ws_size,
                              hipStream_t stream) {
    const float* x      = (const float*)d_in[0];
    const float* in_w   = (const float*)d_in[1];
    const float* gate_w = (const float*)d_in[2];
    const float* gate_b = (const float*)d_in[3];
    const float* data_w = (const float*)d_in[4];
    const float* out_w  = (const float*)d_in[5];
    float*       out    = (float*)d_out;
    _Float16*    ws     = (_Float16*)d_ws;

    // pack weights into f16 fragment layouts + zero the prob accumulator slot
    prep_pack<<<(WTOTAL + 255) / 256, 256, 0, stream>>>(in_w, gate_w, data_w, ws, out);

    // 2 waves/block * 16 rows/wave = 32 batch rows per block
    routenet_fwd<<<BATCH / 32, 64, 0, stream>>>(x, gate_b, out_w, ws, out);
}
